// LSTMSequenceArithmetic_33062658244900
// MI455X (gfx1250) — matrix operand order repaired and checked by reading, not compiled, not run on hardware
//
#include <hip/hip_runtime.h>
#include <hip/hip_bf16.h>

// ---------------------------------------------------------------------------
// 2-layer LSTM digit-addition scan on gfx1250 (MI455X).
// One wave32 owns a 16-sample batch tile for all 64 timesteps.
// All recurrent matmuls are v_wmma_f32_16x16x32_f16 (M=16 samples,
// N=16 gates per tile, K=32). Weights converted to f16 WMMA-friendly
// LDS layout once per block. c-state lives in f32 VGPRs; h/x staged in LDS.
// ---------------------------------------------------------------------------

typedef __attribute__((ext_vector_type(16))) _Float16 v16h;
typedef __attribute__((ext_vector_type(8)))  _Float16 v8h;
typedef __attribute__((ext_vector_type(8)))  float    v8f;

static constexpr int L    = 64;
static constexpr int E    = 8;
static constexpr int H    = 32;   // hidden
static constexpr int G4H  = 128;  // 4*H gates
static constexpr int DIN  = 18;   // E + E + 2

static __device__ inline v8f wmma_f16(v16h a, v16h b, v8f c) {
  return __builtin_amdgcn_wmma_f32_16x16x32_f16(
      /*neg_a=*/false, a, /*neg_b=*/false, b,
      /*c_mod=*/(short)0, c, /*reuse_a=*/false, /*reuse_b=*/false);
}

// Fast activations (v_exp_f32 + v_rcp_f32 paths).
static __device__ inline float fsig(float x) {
  return __builtin_amdgcn_rcpf(1.0f + __expf(-x));
}
static __device__ inline float ftanh_(float x) {
  return 1.0f - 2.0f * __builtin_amdgcn_rcpf(1.0f + __expf(2.0f * x));
}

// B operand (K=32 x N=16). Weight stored in LDS as [gate][K=32] f16.
// Lane n (col); half=0 -> K(0-7,16-23), half=1 -> K(8-15,24-31).
static __device__ inline v16h ldB(const _Float16* w, int row0, int n, int half) {
  const v8h* row = (const v8h*)(w + (size_t)(row0 + n) * 32);
  v8h lo = row[half];
  v8h hi = row[half + 2];
  return __builtin_shufflevector(lo, hi, 0, 1, 2, 3, 4, 5, 6, 7,
                                         8, 9, 10, 11, 12, 13, 14, 15);
}

// A operand (M=16 x K=32) from [M][K] f16 staging tile.
// Lane row = n; half=0 -> K(0-7,16-23), half=1 -> K(8-15,24-31).
static __device__ inline v16h ldA(const _Float16* st, int n, int half) {
  const v8h* row = (const v8h*)(st + (size_t)n * 32);
  v8h lo = row[half];
  v8h hi = row[half + 2];
  return __builtin_shufflevector(lo, hi, 0, 1, 2, 3, 4, 5, 6, 7,
                                         8, 9, 10, 11, 12, 13, 14, 15);
}

// One LSTM layer for one wave-tile: gates = ax*Wih^T + ah*Whh^T + bias,
// then elementwise cell update; writes new h (f16) into hstage [16][32].
static __device__ inline void lstm_layer(v16h ax, v16h ah,
                                         const _Float16* wih,
                                         const _Float16* whh,
                                         const float* biasreg,
                                         float* ca, float* cb,
                                         _Float16* hstage,
                                         int n, int half) {
  v8f acc[8];
#pragma unroll
  for (int tt = 0; tt < 8; ++tt) {
    v8f c;
#pragma unroll
    for (int i = 0; i < 8; ++i) c[i] = biasreg[tt];
    c = wmma_f16(ax, ldB(wih, 16 * tt, n, half), c);
    c = wmma_f16(ah, ldB(whh, 16 * tt, n, half), c);
    acc[tt] = c;
  }
  // Gate tiles: t0/t1 = i, t2/t3 = f, t4/t5 = g, t6/t7 = o.
  // Lane (n,half) + reg r holds hidden units n (even tiles) and 16+n (odd).
#pragma unroll
  for (int r = 0; r < 8; ++r) {
    float i0 = acc[0][r], i1 = acc[1][r];
    float f0 = acc[2][r], f1 = acc[3][r];
    float g0 = acc[4][r], g1 = acc[5][r];
    float o0 = acc[6][r], o1 = acc[7][r];
    ca[r] = fsig(f0) * ca[r] + fsig(i0) * ftanh_(g0);
    cb[r] = fsig(f1) * cb[r] + fsig(i1) * ftanh_(g1);
    float hA = fsig(o0) * ftanh_(ca[r]);
    float hB = fsig(o1) * ftanh_(cb[r]);
    int m = half * 8 + r;
    hstage[(size_t)m * 32 + n]      = (_Float16)hA;
    hstage[(size_t)m * 32 + 16 + n] = (_Float16)hB;
  }
}

struct alignas(64) Smem {
  alignas(64) _Float16 wih0[G4H * 32];   // [gate][K] K padded 18->32
  alignas(64) _Float16 whh0[G4H * 32];
  alignas(64) _Float16 wih1[G4H * 32];
  alignas(64) _Float16 whh1[G4H * 32];
  alignas(64) _Float16 whead[16 * 32];   // rows: 10 digit + 2 carry + 4 zero
  alignas(64) _Float16 ea[10 * E];
  alignas(64) _Float16 eb[10 * E];
  alignas(64) _Float16 xs[4][16 * 32];   // per-wave x staging [M][K]
  alignas(64) _Float16 h0s[4][16 * 32];
  alignas(64) _Float16 h1s[4][16 * 32];
  float bias0[G4H];
  float bias1[G4H];
  float biash[16];
};

__global__ __launch_bounds__(128)
void lstm_seq_kernel(const int* __restrict__ a, const int* __restrict__ b,
                     const float* __restrict__ Ea, const float* __restrict__ Eb,
                     const float* __restrict__ Wih0, const float* __restrict__ Whh0,
                     const float* __restrict__ bih0, const float* __restrict__ bhh0,
                     const float* __restrict__ Wih1, const float* __restrict__ Whh1,
                     const float* __restrict__ bih1, const float* __restrict__ bhh1,
                     const float* __restrict__ Wd, const float* __restrict__ bd,
                     const float* __restrict__ Wc, const float* __restrict__ bc,
                     float* __restrict__ dig_out, float* __restrict__ car_out) {
  __shared__ Smem s;
  const int tid = threadIdx.x;

  // ---- Prologue: convert weights to f16 LDS layouts (block-cooperative) ----
  for (int i = tid; i < G4H * 32; i += 128) {
    int g = i >> 5, k = i & 31;
    s.wih0[i] = (_Float16)(k < DIN ? Wih0[g * DIN + k] : 0.0f);
    s.whh0[i] = (_Float16)Whh0[i];
    s.wih1[i] = (_Float16)Wih1[i];
    s.whh1[i] = (_Float16)Whh1[i];
  }
  for (int i = tid; i < 16 * 32; i += 128) {
    int g = i >> 5, k = i & 31;
    float v = 0.0f;
    if (g < 10)      v = Wd[g * H + k];
    else if (g < 12) v = Wc[(g - 10) * H + k];
    s.whead[i] = (_Float16)v;
  }
  { // tid is always < 128
    s.bias0[tid] = bih0[tid] + bhh0[tid];
    s.bias1[tid] = bih1[tid] + bhh1[tid];
  }
  if (tid < 16) {
    float v = 0.0f;
    if (tid < 10)      v = bd[tid];
    else if (tid < 12) v = bc[tid - 10];
    s.biash[tid] = v;
  }
  if (tid < 10 * E) {
    s.ea[tid] = (_Float16)Ea[tid];
    s.eb[tid] = (_Float16)Eb[tid];
  }
  for (int i = tid; i < 4 * 512; i += 128) {
    int w = i >> 9, j = i & 511;
    // x: zero except K=16 (carry0) = 1.0 initial carry state [1,0]
    s.xs[w][j]  = (_Float16)(((j & 31) == 16) ? 1.0f : 0.0f);
    s.h0s[w][j] = (_Float16)0.0f;
    s.h1s[w][j] = (_Float16)0.0f;
  }
  __syncthreads();

  // ---- Per-wave setup -----------------------------------------------------
  const int lane = tid & 31;
  const int half = lane >> 4;   // 0: K/M-low half lanes, 1: high half
  const int n    = lane & 15;   // column / row-within-tile index
  const int wave = tid >> 5;
  const int sampleBase = blockIdx.x * 64 + wave * 16;

  _Float16* xw  = s.xs[wave];
  _Float16* h0w = s.h0s[wave];
  _Float16* h1w = s.h1s[wave];

  float b0r[8], b1r[8];
#pragma unroll
  for (int tt = 0; tt < 8; ++tt) {
    b0r[tt] = s.bias0[tt * 16 + n];
    b1r[tt] = s.bias1[tt * 16 + n];
  }
  const float bh = s.biash[n];

  float c0a[8], c0b[8], c1a[8], c1b[8];
#pragma unroll
  for (int r = 0; r < 8; ++r) { c0a[r] = c0b[r] = c1a[r] = c1b[r] = 0.0f; }

  // Embeddings for t=0 into x staging. Lanes 0-15: Ea row n; 16-31: Eb row n.
  {
    const int idx0 = (half ? b : a)[(size_t)(sampleBase + n) * L + 0];
    const _Float16* etab = half ? s.eb : s.ea;
    *(v8h*)(xw + (size_t)n * 32 + half * 8) = *(const v8h*)(etab + idx0 * E);
  }

  // ---- Time loop ----------------------------------------------------------
  for (int t = 0; t < L; ++t) {
    // Prefetch next step's digit index early (overlaps WMMA work).
    const int tn = (t + 1) & (L - 1);
    const int idxN = (half ? b : a)[(size_t)(sampleBase + n) * L + tn];

    // Layer 0: gates = x*Wih0^T + h0*Whh0^T
    {
      v16h ax  = ldA(xw, n, half);
      v16h ah0 = ldA(h0w, n, half);
      lstm_layer(ax, ah0, s.wih0, s.whh0, b0r, c0a, c0b, h0w, n, half);
    }
    // Layer 1: gates = h0_new*Wih1^T + h1*Whh1^T
    {
      v16h ah0 = ldA(h0w, n, half);
      v16h ah1 = ldA(h1w, n, half);
      lstm_layer(ah0, ah1, s.wih1, s.whh1, b1r, c1a, c1b, h1w, n, half);
    }
    // Head: logits = h1_new * [Wd;Wc]^T + [bd;bc]
    v8f hc;
#pragma unroll
    for (int i = 0; i < 8; ++i) hc[i] = bh;
    {
      v16h ah1 = ldA(h1w, n, half);
      hc = wmma_f16(ah1, ldB(s.whead, 0, n, half), hc);
    }

    // Exchange carry logits between lanes n=10 <-> n=11 (for softmax).
    float oth[8];
#pragma unroll
    for (int r = 0; r < 8; ++r) oth[r] = __shfl_xor(hc[r], 1, 32);

    if (n < 10) {
#pragma unroll
      for (int r = 0; r < 8; ++r) {
        size_t smp = (size_t)(sampleBase + half * 8 + r);
        dig_out[smp * (L * 10) + (size_t)t * 10 + n] = hc[r];
      }
    } else if (n < 12) {
      const int ci = n - 10;
#pragma unroll
      for (int r = 0; r < 8; ++r) {
        size_t smp = (size_t)(sampleBase + half * 8 + r);
        car_out[smp * (L * 2) + (size_t)t * 2 + ci] = hc[r];
        // softmax over the two carry logits; this lane's probability:
        float x0 = hc[r], x1 = oth[r];
        float mx = fmaxf(x0, x1);
        float e0 = __expf(x0 - mx), e1 = __expf(x1 - mx);
        float p  = e0 * __builtin_amdgcn_rcpf(e0 + e1);
        xw[(size_t)(half * 8 + r) * 32 + 16 + ci] = (_Float16)p;
      }
    }

    // Stage embeddings for t+1 (K 18..31 of x remain zero forever).
    if (t < L - 1) {
      const _Float16* etab = half ? s.eb : s.ea;
      *(v8h*)(xw + (size_t)n * 32 + half * 8) = *(const v8h*)(etab + idxN * E);
    }
  }
}

extern "C" void kernel_launch(void* const* d_in, const int* in_sizes, int n_in,
                              void* d_out, int out_size, void* d_ws, size_t ws_size,
                              hipStream_t stream) {
  const int*   a    = (const int*)  d_in[0];
  const int*   b    = (const int*)  d_in[1];
  const float* Ea   = (const float*)d_in[2];
  const float* Eb   = (const float*)d_in[3];
  const float* Wih0 = (const float*)d_in[4];
  const float* Whh0 = (const float*)d_in[5];
  const float* bih0 = (const float*)d_in[6];
  const float* bhh0 = (const float*)d_in[7];
  const float* Wih1 = (const float*)d_in[8];
  const float* Whh1 = (const float*)d_in[9];
  const float* bih1 = (const float*)d_in[10];
  const float* bhh1 = (const float*)d_in[11];
  const float* Wd   = (const float*)d_in[12];
  const float* bd   = (const float*)d_in[13];
  const float* Wc   = (const float*)d_in[14];
  const float* bc   = (const float*)d_in[15];

  const int Btot = in_sizes[0] / L;          // a is [B, L]
  float* dig = (float*)d_out;                // [B, L, 10]
  float* car = dig + (size_t)Btot * L * 10;  // [B, L, 2]

  const int blocks = Btot / 64;              // 64 samples (4 wave-tiles) per block
  lstm_seq_kernel<<<blocks, 128, 0, stream>>>(
      a, b, Ea, Eb, Wih0, Whh0, bih0, bhh0,
      Wih1, Whh1, bih1, bhh1, Wd, bd, Wc, bc, dig, car);
}
